// LongFormerSelfAttentionBlock_70325794505092
// MI455X (gfx1250) — compile-verified, matrix operand order checked by reading, not run on hardware
//
#include <hip/hip_runtime.h>
#include <hip/hip_bf16.h>

typedef __attribute__((ext_vector_type(16))) _Float16 v16h;
typedef __attribute__((ext_vector_type(8)))  _Float16 v8h;
typedef __attribute__((ext_vector_type(8)))  float    v8f;

// ---------------------------------------------------------------------------
// WMMA helpers (CDNA5 gfx1250, wave32). D = A(16x32 f16) * B(32x16 f16) + C(f32)
// ---------------------------------------------------------------------------
__device__ __forceinline__ v8f wmma_f16(v16h a, v16h b, v8f c) {
  return __builtin_amdgcn_wmma_f32_16x16x32_f16(
      /*neg_a=*/false, a, /*neg_b=*/false, b,
      /*c_mod=*/(short)0, c, /*reuse_a=*/false, /*reuse_b=*/false);
}

// Fragment load, contiguous per lane (2 x 16B -> ds_load_b128 pairs).
// A-matrix 16x32 layout: lane l(0..15): row M=l, K 0..7 & 16..23 ;
//                        lane l+16   : row M=l, K 8..15 & 24..31.
// The same pattern serves as a B(32x16) fragment when the source tile is
// stored K-contiguous per column (i.e. transposed to [N][K]).
__device__ __forceinline__ v16h frag_ld(const _Float16* p, int ld) {
  const int lane = threadIdx.x & 31;
  const _Float16* row = p + (lane & 15) * ld + ((lane >> 4) << 3);
  v16h a;
  *reinterpret_cast<v8h*>(&a) = *reinterpret_cast<const v8h*>(row);
  v8h hi = *reinterpret_cast<const v8h*>(row + 16);
#pragma unroll
  for (int i = 0; i < 8; ++i) a[8 + i] = hi[i];
  return a;
}

// CDNA5 async global->LDS copy, 16 bytes per lane, tracked by ASYNCcnt.
// VDST carries the LDS byte offset (low 32 bits of the flat LDS address,
// per the aperture rules in CDNA5 ISA).
__device__ __forceinline__ void async_copy16(const void* gsrc, unsigned lds_off) {
  asm volatile("global_load_async_to_lds_b128 %0, %1, off"
               :: "v"(lds_off), "v"(gsrc) : "memory");
}
// Async ops complete in order -> waiting to <=4 retires the previous tile's
// 4 copies while the next tile's 4 stay in flight (software pipelining).
__device__ __forceinline__ void wait_async0() {
  asm volatile("s_wait_asynccnt 0x0" ::: "memory");
}
__device__ __forceinline__ void wait_async4() {
  asm volatile("s_wait_asynccnt 0x4" ::: "memory");
}
__device__ __forceinline__ unsigned lds_off(const void* p) {
  return (unsigned)(size_t)p;
}

// ---------------------------------------------------------------------------
// Conversion / padding kernels
// ---------------------------------------------------------------------------
// dst[n*K + k] = (f16) src[k*N + n]   (weight transpose + f32->f16)
__global__ __launch_bounds__(256) void cvt_f16_t_kernel(const float* __restrict__ src,
                                                        _Float16* __restrict__ dst,
                                                        int K, int N) {
  size_t i = (size_t)blockIdx.x * 256 + threadIdx.x;
  if (i >= (size_t)K * N) return;
  int k = (int)(i % K);
  int n = (int)(i / K);
  dst[i] = (_Float16)src[(size_t)k * N + n];
}

// x [B,S,768] f32 -> xp_h [B,Sp,768] f16, zero padded rows
__global__ __launch_bounds__(256) void pad_x_kernel(const float* __restrict__ x,
                                                    _Float16* __restrict__ xp,
                                                    int S, int Sp) {
  size_t i = (size_t)blockIdx.x * 256 + threadIdx.x;
  size_t total = (size_t)4 * Sp * 768;
  if (i >= total) return;
  int d = (int)(i % 768);
  size_t r = i / 768;
  int pos = (int)(r % Sp);
  int b = (int)(r / Sp);
  float v = 0.f;
  if (pos < S) v = x[((size_t)b * S + pos) * 768 + d];
  xp[i] = (_Float16)v;
}

// ---------------------------------------------------------------------------
// Generic f16 WMMA GEMM with TRANSPOSED weights: C[M,N] = A[M,K] * Bt[N,K]^T.
//   op==0 : outH = f16((acc+bias)*scale)            (QKV projections)
//   op==1 : outH = f16(gelu_exact(acc+bias))        (MLP layer 1)
//   op==2 : outF = acc + bias + res                 (MLP layer 2 + residual)
// M,N multiples of 128; K multiple of 32. 256 thr = 8 waves, 128x128 tile.
// Double-buffered LDS; tiles staged with async global->LDS b128 and retired
// with a pipelined s_wait_asynccnt 0x4 (next tile stays in flight).
// ---------------------------------------------------------------------------
__global__ __launch_bounds__(256)
void gemm_f16_kernel(const _Float16* __restrict__ A, const _Float16* __restrict__ Bt,
                     const float* __restrict__ bias, _Float16* __restrict__ outH,
                     float* __restrict__ outF, const float* __restrict__ res,
                     int M, int N, int K, float scale, int op) {
  __shared__ _Float16 As[2][128 * 40];   // 128(M) x 32(K), padded ld=40, x2 buffers
  __shared__ _Float16 Bs[2][128 * 40];   // 128(N) x 32(K), padded ld=40, x2 buffers
  const int tid  = threadIdx.x;
  const int lane = tid & 31;
  const int wv   = tid >> 5;
  const int wm   = wv & 1;            // 2 wave-rows of 64
  const int wn   = wv >> 1;           // 4 wave-cols of 32
  const int m0   = blockIdx.y * 128;
  const int n0   = blockIdx.x * 128;
  const unsigned asb0 = lds_off(As[0]), asb1 = lds_off(As[1]);
  const unsigned bsb0 = lds_off(Bs[0]), bsb1 = lds_off(Bs[1]);

  // per-thread source row/col for staging (2 x 16B vectors per tile per matrix)
  const int row0 = (tid * 2) >> 2,      cv0 = (tid * 2) & 3;
  const int row1 = (tid * 2 + 1) >> 2,  cv1 = (tid * 2 + 1) & 3;
  const unsigned d0 = (unsigned)(row0 * 40 + cv0 * 8) * 2u;
  const unsigned d1 = (unsigned)(row1 * 40 + cv1 * 8) * 2u;

  auto stage = [&](int k0, int pbuf) {
    const unsigned ab = pbuf ? asb1 : asb0;
    const unsigned bb = pbuf ? bsb1 : bsb0;
    async_copy16(A  + (size_t)(m0 + row0) * K + k0 + cv0 * 8, ab + d0);
    async_copy16(Bt + (size_t)(n0 + row0) * K + k0 + cv0 * 8, bb + d0);
    async_copy16(A  + (size_t)(m0 + row1) * K + k0 + cv1 * 8, ab + d1);
    async_copy16(Bt + (size_t)(n0 + row1) * K + k0 + cv1 * 8, bb + d1);
  };

  v8f acc[4][2] = {};

  stage(0, 0);
  int p = 0;
  for (int k0 = 0; k0 < K; k0 += 32, p ^= 1) {
    const bool hasNext = (k0 + 32 < K);
    if (hasNext) {
      stage(k0 + 32, p ^ 1);           // prefetch next tile into other buffer
      wait_async4();                    // retire current tile only (in-order)
    } else {
      wait_async0();
    }
    if (k0 + 64 < K) {                  // keep L2 warm two tiles ahead
      __builtin_prefetch(A  + (size_t)(m0 + (tid & 127)) * K + k0 + 64, 0, 1);
      __builtin_prefetch(Bt + (size_t)(n0 + (tid & 127)) * K + k0 + 64, 0, 1);
    }
    __syncthreads();

    const _Float16* Ap = As[p];
    const _Float16* Bp = Bs[p];
    v16h af[4], bf[2];
#pragma unroll
    for (int r = 0; r < 4; ++r) af[r] = frag_ld(&Ap[(wm * 64 + r * 16) * 40], 40);
#pragma unroll
    for (int c = 0; c < 2; ++c) bf[c] = frag_ld(&Bp[(wn * 32 + c * 16) * 40], 40);
#pragma unroll
    for (int r = 0; r < 4; ++r)
#pragma unroll
      for (int c = 0; c < 2; ++c) acc[r][c] = wmma_f16(af[r], bf[c], acc[r][c]);
    __syncthreads();
  }

  // epilogue (C layout: vgpr e -> M = base + e, lanes>=16 add 8; N = lane&15)
#pragma unroll
  for (int r = 0; r < 4; ++r) {
#pragma unroll
    for (int c = 0; c < 2; ++c) {
      const int col = n0 + wn * 32 + c * 16 + (lane & 15);
      const int rb  = m0 + wm * 64 + r * 16 + ((lane >> 4) << 3);
      const float bcol = bias[col];
#pragma unroll
      for (int e = 0; e < 8; ++e) {
        const size_t o = (size_t)(rb + e) * N + col;
        const float v = acc[r][c][e] + bcol;
        if (op == 0)      outH[o] = (_Float16)(v * scale);
        else if (op == 1) outH[o] = (_Float16)(0.5f * v * (1.f + erff(v * 0.70710678f)));
        else              outF[o] = v + res[o];
      }
    }
  }
}

// ---------------------------------------------------------------------------
// Sliding-window attention + single global key (last padded position).
// One block per (b, head, chunk, 32-row query sub-tile). 256 thr = 8 waves.
// Scores: 32x384x64 WMMA (B = k^T read K-contiguous), softmax over 385 cols,
// PV: 32x64x384 WMMA with the value window staged TRANSPOSED so the B
// fragment is also K-contiguous (all operands via ds_load_b128).
// ---------------------------------------------------------------------------
__global__ __launch_bounds__(256)
void attn_kernel(const _Float16* __restrict__ qb, const _Float16* __restrict__ kb,
                 const _Float16* __restrict__ vb, const unsigned char* __restrict__ msk,
                 float* __restrict__ attn, int S, int Sp) {
  const int nb   = Sp / 128;
  const int tid  = threadIdx.x;
  const int lane = tid & 31;
  const int wv   = tid >> 5;
  int idx = blockIdx.x;
  const int sub    = idx & 3;  idx >>= 2;
  const int nchunk = idx % nb; idx /= nb;
  const int h = idx % 12;
  const int b = idx / 12;
  const int pos0   = nchunk * 128 + sub * 32;
  const int kstart = nchunk * 128 - 128;

  extern __shared__ char smem[];
  _Float16* q_s  = (_Float16*)smem;               // 32 x 72
  _Float16* kv_s = q_s + 32 * 72;                 // keys [384][72]; later vT [64][392]
  float*    sc_s = (float*)(kv_s + 384 * 72);     // 32 x 400 (cols 0..383 window, 384 global)
  _Float16* p_s  = (_Float16*)(sc_s + 32 * 400);  // 32 x 392 probs f16
  const unsigned qsb = lds_off(q_s);
  const unsigned ksb = lds_off(kv_s);

  // stage q (async, 256 x 16B) -- pre-scaled by 1/sqrt(dh) in QKV epilogue
  {
    const int r = tid >> 3, dv = tid & 7;
    async_copy16(&qb[((size_t)b * Sp + pos0 + r) * 768 + h * 64 + dv * 8],
                 qsb + (unsigned)(r * 72 + dv * 8) * 2u);
  }
  // stage 3w key window (async where in range, zero otherwise)
  for (int i = tid; i < 384 * 8; i += 256) {
    const int r = i >> 3, dv = i & 7;
    const int p = kstart + r;
    if (p >= 0 && p < Sp) {
      async_copy16(&kb[((size_t)b * Sp + p) * 768 + h * 64 + dv * 8],
                   ksb + (unsigned)(r * 72 + dv * 8) * 2u);
    } else {
      v8h z = {};
      *reinterpret_cast<v8h*>(&kv_s[r * 72 + dv * 8]) = z;
    }
  }
  wait_async0();
  __syncthreads();

  // scores = q @ k^T : 2 row-tiles x 24 col-tiles, K=64 (2 wmma steps)
  v16h aq[2][2];
#pragma unroll
  for (int rt = 0; rt < 2; ++rt)
#pragma unroll
    for (int ks = 0; ks < 2; ++ks) aq[rt][ks] = frag_ld(&q_s[rt * 16 * 72 + ks * 32], 72);

#pragma unroll
  for (int ci = 0; ci < 3; ++ci) {
    const int ct = wv * 3 + ci;
    v16h bk0 = frag_ld(&kv_s[ct * 16 * 72 + 0], 72);
    v16h bk1 = frag_ld(&kv_s[ct * 16 * 72 + 32], 72);
#pragma unroll
    for (int rt = 0; rt < 2; ++rt) {
      v8f cacc = {};
      cacc = wmma_f16(aq[rt][0], bk0, cacc);
      cacc = wmma_f16(aq[rt][1], bk1, cacc);
      const int col = ct * 16 + (lane & 15);
      const int rb  = rt * 16 + ((lane >> 4) << 3);
#pragma unroll
      for (int e = 0; e < 8; ++e) sc_s[(rb + e) * 400 + col] = cacc[e];
    }
  }
  // global-key score column (key at position Sp-1)
  if (tid < 32) {
    const size_t bkg = ((size_t)b * Sp + Sp - 1) * 768 + h * 64;
    float g = 0.f;
    for (int d = 0; d < 64; ++d) g += (float)q_s[tid * 72 + d] * (float)kb[bkg + d];
    sc_s[tid * 400 + 384] = g;
  }
  __syncthreads();

  // band + key-validity masking, softmax over 385 cols, probs -> f16
  if (tid < 32) {
    const int row = tid;
    const int qpos = pos0 + row;
    const int rchunk = sub * 32 + row;
    const bool qdead = (qpos >= S) || (msk[(size_t)b * S + qpos] != 0);
    float mx = sc_s[row * 400 + 384];
    for (int c2 = 0; c2 < 384; ++c2) {
      const int p = kstart + c2;
      const int dlt = c2 - 128 - rchunk;
      bool ok = (dlt >= -128) && (dlt <= 128) && (p >= 0) && (p < S);
      if (ok) ok = (msk[(size_t)b * S + p] == 0);
      const float v = ok ? sc_s[row * 400 + c2] : -1e9f;
      sc_s[row * 400 + c2] = v;
      mx = fmaxf(mx, v);
    }
    float sum = 0.f;
    const float eg = expf(sc_s[row * 400 + 384] - mx);
    sum += eg;
    for (int c2 = 0; c2 < 384; ++c2) {
      const float e = expf(sc_s[row * 400 + c2] - mx);
      sc_s[row * 400 + c2] = e;
      sum += e;
    }
    float inv = 1.f / sum;
    if (qdead) inv = 0.f;  // masked queries output zero
    for (int c2 = 0; c2 < 384; ++c2)
      p_s[row * 392 + c2] = (_Float16)(sc_s[row * 400 + c2] * inv);
    sc_s[row * 400 + 384] = eg * inv;  // global-token probability
  }
  __syncthreads();

  // stage value window TRANSPOSED: vT[d][r], 64 x 392 (reuses kv_s region).
  // Global reads stay lane-coalesced (consecutive tid -> consecutive d);
  // LDS writes are contiguous 16B (ds_store_b128).
  _Float16* vT = kv_s;
  for (int i = tid; i < 64 * 48; i += 256) {
    const int d = i & 63;
    const int r0 = (i >> 6) * 8;
    v8h tmp;
#pragma unroll
    for (int j = 0; j < 8; ++j) {
      const int p = kstart + r0 + j;
      _Float16 val = (_Float16)0.f;
      if (p >= 0 && p < Sp) val = vb[((size_t)b * Sp + p) * 768 + h * 64 + d];
      tmp[j] = val;
    }
    *reinterpret_cast<v8h*>(&vT[d * 392 + r0]) = tmp;
  }
  __syncthreads();

  // out = probs @ v : 2x4 tiles, one per wave, K=384 (12 wmma steps)
  const int rt  = wv & 1;
  const int ctv = wv >> 1;
  v8f oacc = {};
#pragma unroll
  for (int ks = 0; ks < 12; ++ks) {
    v16h ap  = frag_ld(&p_s[rt * 16 * 392 + ks * 32], 392);
    v16h bvf = frag_ld(&vT[(ctv * 16) * 392 + ks * 32], 392);
    oacc = wmma_f16(ap, bvf, oacc);
  }
  const int dcol = ctv * 16 + (lane & 15);
  const int rb   = rt * 16 + ((lane >> 4) << 3);
  const float vlast = (float)vb[((size_t)b * Sp + Sp - 1) * 768 + h * 64 + dcol];
#pragma unroll
  for (int e = 0; e < 8; ++e) {
    const int row = rb + e;
    const float pg = sc_s[row * 400 + 384];
    attn[((size_t)b * Sp + pos0 + row) * 768 + h * 64 + dcol] = oacc[e] + pg * vlast;
  }
}

// ---------------------------------------------------------------------------
// h = LayerNorm(x + attn) ; writes f32 (final residual) and f16 (MLP input)
// one block per row (b*S + s), D = 768
// ---------------------------------------------------------------------------
__global__ __launch_bounds__(256)
void ln_kernel(const float* __restrict__ x, const float* __restrict__ attn,
               const float* __restrict__ g, const float* __restrict__ bta,
               float* __restrict__ hF, _Float16* __restrict__ hH, int S, int Sp) {
  const int rowid = blockIdx.x;
  const int b = rowid / S, s = rowid % S;
  const int tid = threadIdx.x;
  const float* xr = x + (size_t)rowid * 768;
  const float* ar = attn + ((size_t)b * Sp + s) * 768;
  __shared__ float red[256];

  float v0[3];
  float lsum = 0.f;
#pragma unroll
  for (int j = 0; j < 3; ++j) {
    const int c = tid + j * 256;
    const float t = xr[c] + ar[c];
    v0[j] = t;
    lsum += t;
  }
  red[tid] = lsum; __syncthreads();
  for (int st = 128; st > 0; st >>= 1) { if (tid < st) red[tid] += red[tid + st]; __syncthreads(); }
  const float mu = red[0] * (1.f / 768.f);
  __syncthreads();
  float ls2 = 0.f;
#pragma unroll
  for (int j = 0; j < 3; ++j) { const float d = v0[j] - mu; ls2 += d * d; }
  red[tid] = ls2; __syncthreads();
  for (int st = 128; st > 0; st >>= 1) { if (tid < st) red[tid] += red[tid + st]; __syncthreads(); }
  const float rstd = rsqrtf(red[0] * (1.f / 768.f) + 1e-5f);
#pragma unroll
  for (int j = 0; j < 3; ++j) {
    const int c = tid + j * 256;
    const float hv = (v0[j] - mu) * rstd * g[c] + bta[c];
    hF[(size_t)rowid * 768 + c] = hv;
    hH[(size_t)rowid * 768 + c] = (_Float16)hv;
  }
}

// ---------------------------------------------------------------------------
// Host launcher
// ---------------------------------------------------------------------------
extern "C" void kernel_launch(void* const* d_in, const int* in_sizes, int n_in,
                              void* d_out, int out_size, void* d_ws, size_t ws_size,
                              hipStream_t stream) {
  (void)in_sizes; (void)n_in; (void)out_size; (void)ws_size;
  const int B = 4, S = 4000, D = 768, Sp = 4096, H = 12;
  const float* x   = (const float*)d_in[0];
  const unsigned char* msk = (const unsigned char*)d_in[1];
  const float* Wq = (const float*)d_in[2];  const float* bq = (const float*)d_in[3];
  const float* Wk = (const float*)d_in[4];  const float* bk = (const float*)d_in[5];
  const float* Wv = (const float*)d_in[6];  const float* bv = (const float*)d_in[7];
  const float* lng = (const float*)d_in[8]; const float* lnb = (const float*)d_in[9];
  const float* W1 = (const float*)d_in[10]; const float* b1 = (const float*)d_in[11];
  const float* W2 = (const float*)d_in[12]; const float* b2 = (const float*)d_in[13];

  char* ws = (char*)d_ws;
  size_t off = 0;
  auto carve = [&](size_t bytes) -> void* {
    off = (off + 255) & ~(size_t)255;
    void* p = ws + off;
    off += bytes;
    return p;
  };
  const size_t MSp = (size_t)B * Sp;  // 16384 padded rows
  const size_t Mr  = (size_t)B * S;   // 16000 real rows

  _Float16* xp_h = (_Float16*)carve(MSp * D * 2);
  _Float16* wqT = (_Float16*)carve((size_t)D * D * 2);      // [N=768][K=768]
  _Float16* wkT = (_Float16*)carve((size_t)D * D * 2);
  _Float16* wvT = (_Float16*)carve((size_t)D * D * 2);
  _Float16* w1T = (_Float16*)carve((size_t)D * 4 * D * 2);  // [N=3072][K=768]
  _Float16* w2T = (_Float16*)carve((size_t)D * 4 * D * 2);  // [N=768][K=3072]
  _Float16* qbuf = (_Float16*)carve(MSp * D * 2);
  _Float16* kbuf = (_Float16*)carve(MSp * D * 2);
  _Float16* vbuf = (_Float16*)carve(MSp * D * 2);
  float*    attnb = (float*)carve(MSp * D * 4);
  float*    hF   = (float*)carve(Mr * D * 4);
  _Float16* hH   = (_Float16*)carve(Mr * D * 2);
  _Float16* mid  = (_Float16*)carve(Mr * 4 * D * 2);

  // 1) conversions (+ weight transposes, done once; amortized over ~16K rows)
  { size_t n = MSp * D;
    pad_x_kernel<<<(unsigned)((n + 255) / 256), 256, 0, stream>>>(x, xp_h, S, Sp); }
  { size_t n = (size_t)D * D;
    cvt_f16_t_kernel<<<(unsigned)((n + 255) / 256), 256, 0, stream>>>(Wq, wqT, D, D);
    cvt_f16_t_kernel<<<(unsigned)((n + 255) / 256), 256, 0, stream>>>(Wk, wkT, D, D);
    cvt_f16_t_kernel<<<(unsigned)((n + 255) / 256), 256, 0, stream>>>(Wv, wvT, D, D); }
  { size_t n = (size_t)D * 4 * D;
    cvt_f16_t_kernel<<<(unsigned)((n + 255) / 256), 256, 0, stream>>>(W1, w1T, D, 4 * D);
    cvt_f16_t_kernel<<<(unsigned)((n + 255) / 256), 256, 0, stream>>>(W2, w2T, 4 * D, D); }

  // 2) QKV projections (q pre-scaled by 1/sqrt(64))
  dim3 blk(256);
  gemm_f16_kernel<<<dim3(D / 128, (unsigned)(MSp / 128)), blk, 0, stream>>>(
      xp_h, wqT, bq, qbuf, nullptr, nullptr, (int)MSp, D, D, 0.125f, 0);
  gemm_f16_kernel<<<dim3(D / 128, (unsigned)(MSp / 128)), blk, 0, stream>>>(
      xp_h, wkT, bk, kbuf, nullptr, nullptr, (int)MSp, D, D, 1.0f, 0);
  gemm_f16_kernel<<<dim3(D / 128, (unsigned)(MSp / 128)), blk, 0, stream>>>(
      xp_h, wvT, bv, vbuf, nullptr, nullptr, (int)MSp, D, D, 1.0f, 0);

  // 3) sliding-window + global-token attention
  const unsigned attnBlocks = (unsigned)(B * H * (Sp / 128) * 4);
  const size_t attnSmem = (size_t)(32 * 72 + 384 * 72) * 2 + (size_t)32 * 400 * 4 +
                          (size_t)32 * 392 * 2;  // ~133 KB dynamic LDS (<= 320 KB/WGP)
  attn_kernel<<<attnBlocks, blk, (unsigned)attnSmem, stream>>>(
      qbuf, kbuf, vbuf, msk, attnb, S, Sp);

  // 4) residual + LayerNorm
  ln_kernel<<<(unsigned)Mr, blk, 0, stream>>>(x, attnb, lng, lnb, hF, hH, S, Sp);

  // 5) MLP: mid = gelu(h @ W1 + b1)
  gemm_f16_kernel<<<dim3(4 * D / 128, (unsigned)(Mr / 128)), blk, 0, stream>>>(
      hH, w1T, b1, mid, nullptr, nullptr, (int)Mr, 4 * D, D, 1.0f, 1);

  // 6) out = h + mid @ W2 + b2   (f32 output)
  gemm_f16_kernel<<<dim3(D / 128, (unsigned)(Mr / 128)), blk, 0, stream>>>(
      mid, w2T, b2, nullptr, (float*)d_out, hF, (int)Mr, D, 4 * D, 1.0f, 2);
}
